// PointNetSetAbstraction_81303730913468
// MI455X (gfx1250) — matrix operand order repaired and checked by reading, not compiled, not run on hardware
//
#include <hip/hip_runtime.h>

// PointNet++ Set Abstraction for MI455X (gfx1250, wave32, WMMA).
// Phases: (1) FPS  (2) ball query + relative coords  (3) 3-layer MLP via
// v_wmma_f32_16x16x32_f16 + max-pool over samples.
//
// MLP fragment gathers use contiguous-per-lane LDS layouts so each WMMA
// operand is 2x ds_load_b128 (instead of 16x ds_load_u16) and each D store
// is 1x ds_store_b128.
//
// Sizes fixed by the harness setup: B=8, N=8192, S=2048, NS=32.

typedef __attribute__((ext_vector_type(16))) _Float16 v16h;
typedef __attribute__((ext_vector_type(8)))  _Float16 v8h;
typedef __attribute__((ext_vector_type(8)))  float    v8f;

constexpr int B  = 8;
constexpr int N  = 8192;
constexpr int S  = 2048;      // 0.25 * N
constexpr int NS = 32;        // nsample
constexpr float R2 = 0.2f * 0.2f;
constexpr long OUT2_OFF = (long)B * 3 * S;   // offset of new_points in d_out

// ---------------------------------------------------------------------------
// Kernel 1: farthest point sampling. One block per batch, 1024 threads,
// 8 points per thread held in registers. Writes new_xyz^T (B,3,S) to d_out.
// ---------------------------------------------------------------------------
__global__ __launch_bounds__(1024) void fps_kernel(const float* __restrict__ xyz,
                                                   float* __restrict__ out) {
  const int b = blockIdx.x;
  const int t = threadIdx.x;
  const float* xb = xyz + (long)b * N * 3;

  constexpr int PPT = N / 1024;  // 8
  float px[PPT], py[PPT], pz[PPT], dist[PPT];
#pragma unroll
  for (int i = 0; i < PPT; ++i) {
    int pt = i * 1024 + t;
    px[i] = xb[pt * 3 + 0];
    py[i] = xb[pt * 3 + 1];
    pz[i] = xb[pt * 3 + 2];
    dist[i] = 1e10f;
  }

  __shared__ int   sf;
  __shared__ float swv[32];
  __shared__ int   swi[32];
  if (t == 0) sf = 0;
  __syncthreads();

  for (int s = 0; s < S; ++s) {
    int f = sf;
    float cx = xb[f * 3 + 0];
    float cy = xb[f * 3 + 1];
    float cz = xb[f * 3 + 2];
    if (t == 0) {
      out[(long)b * 3 * S + 0 * S + s] = cx;
      out[(long)b * 3 * S + 1 * S + s] = cy;
      out[(long)b * 3 * S + 2 * S + s] = cz;
    }

    float bv = -1.0f;
    int   bi = 0x7fffffff;
#pragma unroll
    for (int i = 0; i < PPT; ++i) {
      float dx = px[i] - cx, dy = py[i] - cy, dz = pz[i] - cz;
      float d  = dx * dx + dy * dy + dz * dz;
      dist[i]  = fminf(dist[i], d);
      int pt = i * 1024 + t;
      if (dist[i] > bv || (dist[i] == bv && pt < bi)) { bv = dist[i]; bi = pt; }
    }
    // wave32 reduce (argmax, first-occurrence tie-break)
#pragma unroll
    for (int off = 16; off > 0; off >>= 1) {
      float ov = __shfl_down(bv, off);
      int   oi = __shfl_down(bi, off);
      if (ov > bv || (ov == bv && oi < bi)) { bv = ov; bi = oi; }
    }
    if ((t & 31) == 0) { swv[t >> 5] = bv; swi[t >> 5] = bi; }
    __syncthreads();
    if (t < 32) {
      bv = swv[t]; bi = swi[t];
#pragma unroll
      for (int off = 16; off > 0; off >>= 1) {
        float ov = __shfl_down(bv, off);
        int   oi = __shfl_down(bi, off);
        if (ov > bv || (ov == bv && oi < bi)) { bv = ov; bi = oi; }
      }
      if (t == 0) sf = bi;
    }
    __syncthreads();
  }
}

// ---------------------------------------------------------------------------
// Kernel 2: ball query + relative coords. One thread per query point; xyz
// staged through LDS in 2048-point chunks; early exit at 32 neighbors.
// Next chunk is prefetched (global_prefetch_b8) while scanning the current.
// rel layout: rel[((b*S+s)*3 + c)*32 + j]
// ---------------------------------------------------------------------------
constexpr int CHUNK = 2048;

__global__ __launch_bounds__(256) void ball_kernel(const float* __restrict__ xyz,
                                                   const float* __restrict__ out, // new_xyz^T
                                                   float* __restrict__ rel) {
  const int t = threadIdx.x;
  const int blocksPerBatch = S / 256;
  const int b = blockIdx.x / blocksPerBatch;
  const int s = (blockIdx.x % blocksPerBatch) * 256 + t;

  const float* xb = xyz + (long)b * N * 3;
  const float qx = out[(long)b * 3 * S + 0 * S + s];
  const float qy = out[(long)b * 3 * S + 1 * S + s];
  const float qz = out[(long)b * 3 * S + 2 * S + s];

  __shared__ float sx[CHUNK], sy[CHUNK], sz[CHUNK];

  float* relp = rel + ((long)(b * S + s)) * 3 * NS;
  int cnt = 0;
  float r0x = 0.f, r0y = 0.f, r0z = 0.f;

  for (int c0 = 0; c0 < N; c0 += CHUNK) {
    for (int i = t; i < CHUNK; i += 256) {
      int p = c0 + i;
      sx[i] = xb[p * 3 + 0];
      sy[i] = xb[p * 3 + 1];
      sz[i] = xb[p * 3 + 2];
    }
    // prefetch next chunk while this one is scanned (global_prefetch_b8)
    if (c0 + CHUNK < N) {
      for (int i = t; i < CHUNK; i += 256)
        __builtin_prefetch(&xb[(c0 + CHUNK + i) * 3], 0, 1);
    }
    __syncthreads();
    if (cnt < NS) {
      for (int i = 0; i < CHUNK && cnt < NS; ++i) {
        float dx = sx[i] - qx, dy = sy[i] - qy, dz = sz[i] - qz;
        float d  = dx * dx + dy * dy + dz * dz;
        if (d <= R2) {
          relp[0 * NS + cnt] = dx;
          relp[1 * NS + cnt] = dy;
          relp[2 * NS + cnt] = dz;
          if (cnt == 0) { r0x = dx; r0y = dy; r0z = dz; }
          ++cnt;
        }
      }
    }
    __syncthreads();
  }
  if (cnt == 0) {  // JAX OOB-gather clamps to last point
    r0x = xb[(N - 1) * 3 + 0] - qx;
    r0y = xb[(N - 1) * 3 + 1] - qy;
    r0z = xb[(N - 1) * 3 + 2] - qz;
  }
  for (int j = cnt; j < NS; ++j) {
    relp[0 * NS + j] = r0x;
    relp[1 * NS + j] = r0y;
    relp[2 * NS + j] = r0z;
  }
}

// ---------------------------------------------------------------------------
// Kernel 3: MLP (3->64->64->128, ReLU) via v_wmma_f32_16x16x32_f16 + maxpool.
// Block = 256 threads (8 waves), 64 pixels (2 query pts x 32 samples).
// Weights stored [m][Kpad]; activations stored [pixel][Kpad] so that every
// WMMA A/B fragment is two aligned 16-byte LDS loads and every D store is
// one aligned 16-byte LDS store (ISA 7.12.2 16-bit A/B + f32 C/D layouts).
// ---------------------------------------------------------------------------
constexpr int NP = 64;  // pixels per block

__device__ inline v16h load_frag_a(const _Float16* W, int mbase, int kc, int Kpad, int lane) {
  int row = lane & 15;
  int kb  = (lane >> 4) << 3;   // upper half-wave: +8
  const _Float16* p = W + (mbase + row) * Kpad + kc * 32 + kb;
  v8h lo = *(const v8h*)(p);        // K = kb .. kb+7      (VGPR 0..3)
  v8h hi = *(const v8h*)(p + 16);   // K = kb+16 .. kb+23  (VGPR 4..7)
  return __builtin_shufflevector(lo, hi, 0, 1, 2, 3, 4, 5, 6, 7,
                                 8, 9, 10, 11, 12, 13, 14, 15);
}

__device__ inline v16h load_frag_b(const _Float16* X, int nbase, int kc, int Kpad, int lane) {
  int col = lane & 15;
  int kb  = (lane >> 4) << 3;   // upper half-wave: +8
  const _Float16* p = X + (nbase + col) * Kpad + kc * 32 + kb;
  v8h lo = *(const v8h*)(p);        // K = kb .. kb+7
  v8h hi = *(const v8h*)(p + 16);   // K = kb+16 .. kb+23
  return __builtin_shufflevector(lo, hi, 0, 1, 2, 3, 4, 5, 6, 7,
                                 8, 9, 10, 11, 12, 13, 14, 15);
}

__device__ inline void store_relu(_Float16* Xo, const v8f& acc, const float* bias,
                                  int mbase, int nbase, int Mtot, int lane) {
  int col   = lane & 15;
  int rbase = (lane >> 4) << 3;
  v8h d;
#pragma unroll
  for (int v = 0; v < 8; ++v) {
    int m = mbase + rbase + v;
    d[v] = (_Float16)fmaxf(acc[v] + bias[m], 0.0f);
  }
  *(v8h*)(Xo + (nbase + col) * Mtot + mbase + rbase) = d;   // ds_store_b128
}

__global__ __launch_bounds__(256) void mlp_kernel(const float* __restrict__ rel,
                                                  const float* __restrict__ w1,
                                                  const float* __restrict__ b1,
                                                  const float* __restrict__ w2,
                                                  const float* __restrict__ b2,
                                                  const float* __restrict__ w3,
                                                  const float* __restrict__ b3,
                                                  float* __restrict__ out) {
  const int t = threadIdx.x;
  const int blocksPerBatch = S / 2;                  // 2 query points per block
  const int b     = blockIdx.x / blocksPerBatch;
  const int qbase = (blockIdx.x % blocksPerBatch) * 2;

  __shared__ _Float16 sW1[64 * 32];    //  4 KB  [m][32]  (K padded 3->32)
  __shared__ _Float16 sW2[64 * 64];    //  8 KB  [m][64]
  __shared__ _Float16 sW3[128 * 64];   // 16 KB  [m][64]
  __shared__ float    sB1[64];
  __shared__ float    sB2[64];
  __shared__ float    sB3[128];
  __shared__ _Float16 sP[NP * 128];    // 16 KB: X0 = [pix][32] (first 2K halves),
                                       //        later X3 = [pix][128]
  __shared__ _Float16 sX1[NP * 64];    //  8 KB  [pix][64]
  __shared__ _Float16 sX2[NP * 64];    //  8 KB  [pix][64]

  // --- stage weights + input tile --------------------------------------
  for (int i = t; i < 64 * 32; i += 256) {
    int o = i >> 5, c = i & 31;
    sW1[i] = (_Float16)(c < 3 ? w1[o * 3 + c] : 0.0f);
  }
  for (int i = t; i < 64 * 64; i += 256)  sW2[i] = (_Float16)w2[i];
  for (int i = t; i < 128 * 64; i += 256) sW3[i] = (_Float16)w3[i];
  if (t < 64)  sB1[t] = b1[t];
  if (t < 64)  sB2[t] = b2[t];
  if (t < 128) sB3[t] = b3[t];

  const long relbase = ((long)(b * S + qbase)) * 3 * NS;   // rel[b][s][c][j]
  for (int i = t; i < NP * 32; i += 256) {                 // X0: [pix][32]
    int p = i >> 5, k = i & 31;
    int ql = p >> 5, j = p & 31;
    float v = (k < 3) ? rel[relbase + (long)ql * 3 * NS + k * NS + j] : 0.0f;
    sP[i] = (_Float16)v;
  }
  __syncthreads();

  const int wave = t >> 5, lane = t & 31;

  // --- layer 1: 3(->32) x 64 -> 64, 16 tiles, 1 wmma each ---------------
#pragma unroll
  for (int r = 0; r < 2; ++r) {
    int tid = wave * 2 + r;            // 0..15
    int mt = tid >> 2, nt = tid & 3;
    v8f acc = {};
    v16h a  = load_frag_a(sW1, mt * 16, 0, 32, lane);
    v16h bb = load_frag_b(sP, nt * 16, 0, 32, lane);
    acc = __builtin_amdgcn_wmma_f32_16x16x32_f16(false, a, false, bb,
                                                 (short)0, acc, false, false);
    store_relu(sX1, acc, sB1, mt * 16, nt * 16, 64, lane);
  }
  __syncthreads();

  // --- layer 2: 64 x 64 -> 64, 16 tiles, 2 wmma each --------------------
#pragma unroll
  for (int r = 0; r < 2; ++r) {
    int tid = wave * 2 + r;
    int mt = tid >> 2, nt = tid & 3;
    v8f acc = {};
#pragma unroll
    for (int kc = 0; kc < 2; ++kc) {
      v16h a  = load_frag_a(sW2, mt * 16, kc, 64, lane);
      v16h bb = load_frag_b(sX1, nt * 16, kc, 64, lane);
      acc = __builtin_amdgcn_wmma_f32_16x16x32_f16(false, a, false, bb,
                                                   (short)0, acc, false, false);
    }
    store_relu(sX2, acc, sB2, mt * 16, nt * 16, 64, lane);
  }
  __syncthreads();

  // --- layer 3: 64 x 64 -> 128, 32 tiles, 2 wmma each -------------------
#pragma unroll
  for (int r = 0; r < 4; ++r) {
    int tid = wave * 4 + r;            // 0..31
    int mt = tid >> 2, nt = tid & 3;   // mt 0..7
    v8f acc = {};
#pragma unroll
    for (int kc = 0; kc < 2; ++kc) {
      v16h a  = load_frag_a(sW3, mt * 16, kc, 64, lane);
      v16h bb = load_frag_b(sX2, nt * 16, kc, 64, lane);
      acc = __builtin_amdgcn_wmma_f32_16x16x32_f16(false, a, false, bb,
                                                   (short)0, acc, false, false);
    }
    store_relu(sP, acc, sB3, mt * 16, nt * 16, 128, lane);  // X3 = [pix][128]
  }
  __syncthreads();

  // --- max-pool over the 32 samples of each query point -----------------
  {
    int qp = t >> 7;          // 0..1
    int ch = t & 127;         // 0..127
    float m = 0.0f;           // post-ReLU values are >= 0
#pragma unroll
    for (int j = 0; j < NS; ++j)
      m = fmaxf(m, (float)sP[(qp * NS + j) * 128 + ch]);
    int s = qbase + qp;
    out[OUT2_OFF + ((long)b * 128 + ch) * S + s] = m;
  }
}

// ---------------------------------------------------------------------------
extern "C" void kernel_launch(void* const* d_in, const int* in_sizes, int n_in,
                              void* d_out, int out_size, void* d_ws, size_t ws_size,
                              hipStream_t stream) {
  const float* xyz = (const float*)d_in[0];
  // d_in[1] = features (unused by the reference graph)
  const float* w1 = (const float*)d_in[2];
  const float* b1 = (const float*)d_in[3];
  const float* w2 = (const float*)d_in[4];
  const float* b2 = (const float*)d_in[5];
  const float* w3 = (const float*)d_in[6];
  const float* b3 = (const float*)d_in[7];
  float* out = (float*)d_out;
  float* rel = (float*)d_ws;   // B*S*3*NS floats = 6 MB scratch

  fps_kernel<<<B, 1024, 0, stream>>>(xyz, out);
  ball_kernel<<<B * (S / 256), 256, 0, stream>>>(xyz, out, rel);
  mlp_kernel<<<B * (S / 2), 256, 0, stream>>>(rel, w1, b1, w2, b2, w3, b3, out);
}
